// GGNNModel_23691039605197
// MI455X (gfx1250) — matrix operand order repaired and checked by reading, not compiled, not run on hardware
//
#include <hip/hip_runtime.h>
#include <hip/hip_bf16.h>

// GGNN: B=8, N=4096, ATTR=16, H=64, OUT=1, STEPS=8
//  - adjacency fp32 -> f16 (exact for 0/1) + transpose, once per launch
//  - per step: pack h into WMMA B-fragment layout; two WMMA matmuls (A@h, A^T@h)
//    with async-to-LDS double-buffered B fragments; fused GRU gates
//  - final projection h @ Wo^T + bo

#define GG_B 8
#define GG_N 4096
#define GG_ATTR 16
#define GG_H 64
#define GG_STEPS 8

typedef __attribute__((ext_vector_type(16))) _Float16 v16h;
typedef __attribute__((ext_vector_type(8)))  _Float16 v8h;
typedef __attribute__((ext_vector_type(8)))  float    v8f;
typedef __attribute__((ext_vector_type(4)))  int      v4i;

typedef __attribute__((address_space(1))) v4i* gptr_v4i;
typedef __attribute__((address_space(3))) v4i* lptr_v4i;

// ---------------------------------------------------------------------------
// h0 = relu(attr @ Wi^T + bi)
// ---------------------------------------------------------------------------
__global__ void __launch_bounds__(256) k_init_h(const float* __restrict__ attr,
                                                const float* __restrict__ Wi,
                                                const float* __restrict__ bi,
                                                float* __restrict__ h32) {
  int t = blockIdx.x * 256 + threadIdx.x;          // over B*N*H
  int hh = t & (GG_H - 1);
  size_t node = (size_t)(t >> 6);
  const float* a = attr + node * GG_ATTR;
  const float* w = Wi + hh * GG_ATTR;
  float acc = bi[hh];
#pragma unroll
  for (int c = 0; c < GG_ATTR; ++c) acc += a[c] * w[c];
  h32[t] = fmaxf(acc, 0.0f);
}

// ---------------------------------------------------------------------------
// A fp32 -> A f16 (same layout) and AT f16 (transposed) via LDS 64x64 tiles
// ---------------------------------------------------------------------------
__global__ void __launch_bounds__(256) k_convert(const float* __restrict__ A,
                                                 _Float16* __restrict__ A16,
                                                 _Float16* __restrict__ AT16) {
  __shared__ _Float16 tile[64 * 66];
  const int c0 = blockIdx.x * 64;
  const int r0 = blockIdx.y * 64;
  const int b  = blockIdx.z;
  const size_t base = (size_t)b * GG_N * GG_N;
#pragma unroll
  for (int i = 0; i < 16; ++i) {
    int idx = threadIdx.x + i * 256;
    int r = idx >> 6, c = idx & 63;
    float v = A[base + (size_t)(r0 + r) * GG_N + (c0 + c)];
    _Float16 hv = (_Float16)v;
    A16[base + (size_t)(r0 + r) * GG_N + (c0 + c)] = hv;
    tile[r * 66 + c] = hv;
  }
  __syncthreads();
#pragma unroll
  for (int i = 0; i < 16; ++i) {
    int idx = threadIdx.x + i * 256;
    int rr = idx >> 6, cc = idx & 63;
    AT16[base + (size_t)(c0 + rr) * GG_N + (r0 + cc)] = tile[cc * 66 + rr];
  }
}

// ---------------------------------------------------------------------------
// Pack h (f32) into WMMA B-fragment layout (f16):
//   group = ((b*(N/32) + k0/32)*4 + ntile); lane 16 halves:
//   elem e -> h[b][k0 + e + 16*(lane>=16)][ntile*16 + (lane&15)]
// ---------------------------------------------------------------------------
__global__ void __launch_bounds__(256) k_pack_h(const float* __restrict__ h32,
                                                _Float16* __restrict__ hB) {
  int t = blockIdx.x * 256 + threadIdx.x;
  int lane = t & 31;
  int group = t >> 5;
  int nt = group & 3;
  int g2 = group >> 2;
  int ktile = g2 & (GG_N / 32 - 1);
  int b = g2 >> 7;
  int hi = (lane >> 4) * 16;
  int l15 = lane & 15;
  const float* src = h32 + ((size_t)b * GG_N + ktile * 32 + hi) * GG_H + nt * 16 + l15;
  v16h v;
#pragma unroll
  for (int e = 0; e < 16; ++e) v[e] = (_Float16)src[(size_t)e * GG_H];
  ((v16h*)hB)[(size_t)group * 32 + lane] = v;
}

// ---------------------------------------------------------------------------
// Stage one 16KB slab (128 K-values of B-fragments) into LDS.
// Uses CDNA5 async global->LDS copies when available.
// ---------------------------------------------------------------------------
__device__ __forceinline__ void stage_slab(const _Float16* __restrict__ src,
                                           _Float16* dst) {
#pragma unroll
  for (int i = 0; i < 4; ++i) {
    int idx = threadIdx.x + i * 256;        // 1024 x 16B = 16KB
#if __has_builtin(__builtin_amdgcn_global_load_async_to_lds_b128)
    __builtin_amdgcn_global_load_async_to_lds_b128(
        (gptr_v4i)(src + idx * 8),
        (lptr_v4i)(dst + idx * 8),
        0, 0);
#else
    *(v4i*)(dst + idx * 8) = *(const v4i*)(src + idx * 8);
#endif
  }
}

__device__ __forceinline__ void wait_slab() {
#if __has_builtin(__builtin_amdgcn_global_load_async_to_lds_b128)
  asm volatile("s_wait_asynccnt 0" ::: "memory");
#endif
  __syncthreads();
}

// ---------------------------------------------------------------------------
// C (BxNxH, f32) = A16 (BxNxN, f16) @ h  via V_WMMA_F32_16X16X32_F16.
// One wave per 32 rows (two 16-row tiles); B fragments double-buffered in LDS.
// ---------------------------------------------------------------------------
__global__ void __launch_bounds__(256) k_spmm(const _Float16* __restrict__ A16,
                                              const _Float16* __restrict__ hB,
                                              float* __restrict__ C) {
  // [buf][kc(4 x 32k)][ntile(4)][lane(32)][16 halves] = 2 x 16KB
  __shared__ _Float16 ldsB[2][4][4][32][16];

  const int wave = threadIdx.x >> 5;
  const int lane = threadIdx.x & 31;
  const int gw = blockIdx.x * 8 + wave;
  const int rg = gw & (GG_N / 32 - 1);      // 32-row group 0..127
  const int b  = gw >> 7;                   // batch (uniform within block)
  const int half = lane >> 4;
  const int l15 = lane & 15;

  const _Float16* Arow0 = A16 + ((size_t)b * GG_N + rg * 32 + l15) * GG_N + 16 * half;
  const _Float16* Arow1 = Arow0 + (size_t)16 * GG_N;
  const _Float16* hBb = hB + (size_t)b * (GG_N / 32) * 2048;   // 2048 halves per 32-k

  v8f acc00 = {}, acc01 = {}, acc02 = {}, acc03 = {};
  v8f acc10 = {}, acc11 = {}, acc12 = {}, acc13 = {};

  const int NCHUNK = GG_N / 128;            // 32 chunks of 128 K
  stage_slab(hBb, &ldsB[0][0][0][0][0]);
  wait_slab();

  for (int c = 0; c < NCHUNK; ++c) {
    const int cur = c & 1;
    if (c + 1 < NCHUNK) {
      stage_slab(hBb + (size_t)(c + 1) * 8192, &ldsB[cur ^ 1][0][0][0][0]);
      __builtin_prefetch(Arow0 + (c + 1) * 128, 0, 0);
      __builtin_prefetch(Arow1 + (c + 1) * 128, 0, 0);
    }
#pragma unroll
    for (int kc = 0; kc < 4; ++kc) {
      const int k0 = c * 128 + kc * 32;
      v8h a0lo = *(const v8h*)(Arow0 + k0);
      v8h a0hi = *(const v8h*)(Arow0 + k0 + 8);
      v8h a1lo = *(const v8h*)(Arow1 + k0);
      v8h a1hi = *(const v8h*)(Arow1 + k0 + 8);
      v16h a0 = __builtin_shufflevector(a0lo, a0hi, 0, 1, 2, 3, 4, 5, 6, 7,
                                        8, 9, 10, 11, 12, 13, 14, 15);
      v16h a1 = __builtin_shufflevector(a1lo, a1hi, 0, 1, 2, 3, 4, 5, 6, 7,
                                        8, 9, 10, 11, 12, 13, 14, 15);
      const _Float16* lb = &ldsB[cur][kc][0][lane][0];
      v16h b0 = *(const v16h*)(lb);
      v16h b1 = *(const v16h*)(lb + 512);
      v16h b2 = *(const v16h*)(lb + 1024);
      v16h b3 = *(const v16h*)(lb + 1536);
      acc00 = __builtin_amdgcn_wmma_f32_16x16x32_f16(false, a0, false, b0, (short)0, acc00, false, false);
      acc01 = __builtin_amdgcn_wmma_f32_16x16x32_f16(false, a0, false, b1, (short)0, acc01, false, false);
      acc02 = __builtin_amdgcn_wmma_f32_16x16x32_f16(false, a0, false, b2, (short)0, acc02, false, false);
      acc03 = __builtin_amdgcn_wmma_f32_16x16x32_f16(false, a0, false, b3, (short)0, acc03, false, false);
      acc10 = __builtin_amdgcn_wmma_f32_16x16x32_f16(false, a1, false, b0, (short)0, acc10, false, false);
      acc11 = __builtin_amdgcn_wmma_f32_16x16x32_f16(false, a1, false, b1, (short)0, acc11, false, false);
      acc12 = __builtin_amdgcn_wmma_f32_16x16x32_f16(false, a1, false, b2, (short)0, acc12, false, false);
      acc13 = __builtin_amdgcn_wmma_f32_16x16x32_f16(false, a1, false, b3, (short)0, acc13, false, false);
    }
    wait_slab();
  }

  // C/D layout: VGPR v, lane -> (M = v + 8*half, Nc = l15)
#pragma unroll
  for (int v = 0; v < 8; ++v) {
    size_t ro0 = ((size_t)b * GG_N + rg * 32 + v + 8 * half) * GG_H + l15;
    C[ro0 + 0]  = acc00[v];
    C[ro0 + 16] = acc01[v];
    C[ro0 + 32] = acc02[v];
    C[ro0 + 48] = acc03[v];
    size_t ro1 = ro0 + (size_t)16 * GG_H;
    C[ro1 + 0]  = acc10[v];
    C[ro1 + 16] = acc11[v];
    C[ro1 + 32] = acc12[v];
    C[ro1 + 48] = acc13[v];
  }
}

// ---------------------------------------------------------------------------
// Fused GRU update: 4 nodes per block, 64 threads per node.
// ---------------------------------------------------------------------------
__global__ void __launch_bounds__(256) k_gru(const float* __restrict__ a_in,
                                             const float* __restrict__ a_out,
                                             float* __restrict__ h32,
                                             const float* __restrict__ Wz, const float* __restrict__ bz,
                                             const float* __restrict__ Wr, const float* __restrict__ br,
                                             const float* __restrict__ Wh, const float* __restrict__ bh) {
  __shared__ float xs[4][3 * GG_H];
  __shared__ float rh[4][GG_H];
  const int ln = threadIdx.x >> 6;
  const int t  = threadIdx.x & 63;
  const size_t node = (size_t)blockIdx.x * 4 + ln;

  xs[ln][t]       = a_in[node * GG_H + t];
  xs[ln][64 + t]  = a_out[node * GG_H + t];
  xs[ln][128 + t] = h32[node * GG_H + t];
  __syncthreads();

  const float* wz = Wz + t * 3 * GG_H;
  const float* wr = Wr + t * 3 * GG_H;
  float accz = bz[t];
  float accr = br[t];
#pragma unroll 4
  for (int c = 0; c < 3 * GG_H; ++c) {
    float x = xs[ln][c];
    accz += x * wz[c];
    accr += x * wr[c];
  }
  float z = 1.0f / (1.0f + expf(-accz));
  float r = 1.0f / (1.0f + expf(-accr));
  rh[ln][t] = r * xs[ln][128 + t];
  __syncthreads();

  const float* wh = Wh + t * 3 * GG_H;
  float acch = bh[t];
#pragma unroll 4
  for (int c = 0; c < 2 * GG_H; ++c) acch += xs[ln][c] * wh[c];
#pragma unroll 4
  for (int c = 0; c < GG_H; ++c) acch += rh[ln][c] * wh[128 + c];
  float hn = tanhf(acch);
  float hv = xs[ln][128 + t];
  h32[node * GG_H + t] = (1.0f - z) * hv + z * hn;
}

// ---------------------------------------------------------------------------
// out = h @ Wo^T + bo  (OUT = 1)
// ---------------------------------------------------------------------------
__global__ void __launch_bounds__(256) k_out(const float* __restrict__ h32,
                                             const float* __restrict__ Wo,
                                             const float* __restrict__ bo,
                                             float* __restrict__ out) {
  int t = blockIdx.x * 256 + threadIdx.x;            // over B*N
  const float* hp = h32 + (size_t)t * GG_H;
  float acc = bo[0];
#pragma unroll 4
  for (int c = 0; c < GG_H; ++c) acc += hp[c] * Wo[c];
  out[t] = acc;
}

// ---------------------------------------------------------------------------
// Host-side orchestration
// ---------------------------------------------------------------------------
extern "C" void kernel_launch(void* const* d_in, const int* in_sizes, int n_in,
                              void* d_out, int out_size, void* d_ws, size_t ws_size,
                              hipStream_t stream) {
  const float* attr = (const float*)d_in[0];
  const float* adj  = (const float*)d_in[1];
  const float* Wi   = (const float*)d_in[2];
  const float* bi   = (const float*)d_in[3];
  const float* Wz   = (const float*)d_in[4];
  const float* bz   = (const float*)d_in[5];
  const float* Wr   = (const float*)d_in[6];
  const float* br   = (const float*)d_in[7];
  const float* Wh   = (const float*)d_in[8];
  const float* bh   = (const float*)d_in[9];
  const float* Wo   = (const float*)d_in[10];
  const float* bo   = (const float*)d_in[11];
  float* out = (float*)d_out;

  char* ws = (char*)d_ws;
  const size_t szA16 = (size_t)GG_B * GG_N * GG_N * sizeof(_Float16);   // 256 MB
  const size_t szH32 = (size_t)GG_B * GG_N * GG_H * sizeof(float);      // 8 MB
  const size_t szHB  = (size_t)GG_B * GG_N * GG_H * sizeof(_Float16);   // 4 MB

  _Float16* A16  = (_Float16*)(ws);
  _Float16* AT16 = (_Float16*)(ws + szA16);
  float*    h32  = (float*)   (ws + 2 * szA16);
  _Float16* hB   = (_Float16*)(ws + 2 * szA16 + szH32);
  float*    a_in = (float*)   (ws + 2 * szA16 + szH32 + szHB);
  float*    a_out= (float*)   (ws + 2 * szA16 + szH32 + szHB + szH32);

  k_convert<<<dim3(GG_N / 64, GG_N / 64, GG_B), 256, 0, stream>>>(adj, A16, AT16);
  k_init_h<<<(GG_B * GG_N * GG_H) / 256, 256, 0, stream>>>(attr, Wi, bi, h32);

  for (int s = 0; s < GG_STEPS; ++s) {
    k_pack_h<<<(GG_B * (GG_N / 32) * 4 * 32) / 256, 256, 0, stream>>>(h32, hB);
    k_spmm<<<(GG_B * (GG_N / 32)) / 8, 256, 0, stream>>>(A16, hB, a_in);
    k_spmm<<<(GG_B * (GG_N / 32)) / 8, 256, 0, stream>>>(AT16, hB, a_out);
    k_gru<<<(GG_B * GG_N) / 4, 256, 0, stream>>>(a_in, a_out, h32,
                                                 Wz, bz, Wr, br, Wh, bh);
  }

  k_out<<<(GG_B * GG_N) / 256, 256, 0, stream>>>(h32, Wo, bo, out);
}
